// MoE_71365176590647
// MI455X (gfx1250) — matrix operand order repaired and checked by reading, not compiled, not run on hardware
//
#include <hip/hip_runtime.h>
#include <hip/hip_bf16.h>
#include <math.h>

#define N_TOK 8192
#define DIM   1024
#define HID   4096
#define NEXP  4

typedef __attribute__((ext_vector_type(16))) __bf16 v16bf;
typedef __attribute__((ext_vector_type(8)))  float  v8f;

union BF16Frag { v16bf v; unsigned u[8]; };

__device__ __forceinline__ unsigned short f32_bf16(float f) {
  unsigned u = __float_as_uint(f);
  u += 0x7FFFu + ((u >> 16) & 1u);          // round-to-nearest-even
  return (unsigned short)(u >> 16);
}
__device__ __forceinline__ unsigned pack_bf16x2(float lo, float hi) {
  return (unsigned)f32_bf16(lo) | ((unsigned)f32_bf16(hi) << 16);
}

// ---------------------------------------------------------------------------
// Kernel 0: one-time weight pre-pack, f32 -> bf16, swizzled into the exact
// WMMA B-fragment order: dst[((rt*kChunks + kk)*32 + lane)*8 + v] holds the
// bf16 pair  src[rt*16 + (lane&15)][kk*32 + (lane<16?0:8) + ((v&4)<<2) + 2(v&3) .. +1].
// ---------------------------------------------------------------------------
__global__ __launch_bounds__(256) void moe_pack_frag(
    const float* __restrict__ src, unsigned* __restrict__ dst,
    int kChunks, long total) {
  long idx = (long)blockIdx.x * 256 + threadIdx.x;
  if (idx >= total) return;
  int  v    = (int)(idx & 7);
  int  lane = (int)((idx >> 3) & 31);
  long rest = idx >> 8;
  int  kk   = (int)(rest % kChunks);
  long rt   = rest / kChunks;
  long C    = (long)kChunks * 32;
  long row  = rt * 16 + (lane & 15);
  long col  = (long)kk * 32 + ((lane < 16) ? 0 : 8) + ((v & 4) << 2) + 2 * (v & 3);
  const float* s = src + row * C + col;
  dst[idx] = pack_bf16x2(s[0], s[1]);
}

// ---------------------------------------------------------------------------
// Kernel 1: noisy top-2 gating.  One wave (32 lanes) per token.
// ---------------------------------------------------------------------------
__global__ __launch_bounds__(256) void moe_gate(
    const float* __restrict__ x, const float* __restrict__ eps,
    const float* __restrict__ Wg, const float* __restrict__ Wn,
    float* __restrict__ combine) {
  const int wave = threadIdx.x >> 5;
  const int lane = threadIdx.x & 31;
  const int n = blockIdx.x * 8 + wave;
  if (n >= N_TOK) return;

  const float* xr = x + (size_t)n * DIM;
  float acc[NEXP + 1] = {0.f, 0.f, 0.f, 0.f, 0.f};
  for (int d = lane * 4; d < DIM; d += 32 * 4) {
    float4 xv = *(const float4*)(xr + d);
#pragma unroll
    for (int e = 0; e < NEXP; ++e) {
      float4 wv = *(const float4*)(Wg + (size_t)e * DIM + d);
      acc[e] += xv.x * wv.x + xv.y * wv.y + xv.z * wv.z + xv.w * wv.w;
    }
    float4 nv = *(const float4*)(Wn + d);
    acc[NEXP] += xv.x * nv.x + xv.y * nv.y + xv.z * nv.z + xv.w * nv.w;
  }
#pragma unroll
  for (int off = 16; off > 0; off >>= 1)
#pragma unroll
    for (int q = 0; q < NEXP + 1; ++q)
      acc[q] += __shfl_xor(acc[q], off, 32);

  if (lane == 0) {
    float nv = acc[NEXP];
    float noise = (nv > 20.f) ? nv : log1pf(__expf(nv));   // softplus
    float h[NEXP];
#pragma unroll
    for (int e = 0; e < NEXP; ++e)
      h[e] = acc[e] + eps[(size_t)n * NEXP + e] * noise;
    int i1 = 0;
#pragma unroll
    for (int e = 1; e < NEXP; ++e) if (h[e] > h[i1]) i1 = e;
    int i2 = -1;
#pragma unroll
    for (int e = 0; e < NEXP; ++e)
      if (e != i1 && (i2 < 0 || h[e] > h[i2])) i2 = e;
    float e2  = __expf(h[i2] - h[i1]);
    float inv = 1.f / (1.f + e2);
    float c[NEXP] = {0.f, 0.f, 0.f, 0.f};
    c[i1] = inv;
    c[i2] = e2 * inv;
    *(float4*)(combine + (size_t)n * NEXP) = make_float4(c[0], c[1], c[2], c[3]);
  }
}

// ---------------------------------------------------------------------------
// Kernel 2: load-balance loss = (unbiased std / mean)^2 of combine col sums.
// ---------------------------------------------------------------------------
__global__ __launch_bounds__(256) void moe_loss(
    const float* __restrict__ combine, float* __restrict__ out_loss) {
  __shared__ float red[256];
  float s[NEXP] = {0.f, 0.f, 0.f, 0.f};
  for (int n = threadIdx.x; n < N_TOK; n += 256) {
    float4 c = *(const float4*)(combine + (size_t)n * NEXP);
    s[0] += c.x; s[1] += c.y; s[2] += c.z; s[3] += c.w;
  }
  float col[NEXP];
#pragma unroll
  for (int e = 0; e < NEXP; ++e) {
    red[threadIdx.x] = s[e];
    __syncthreads();
    for (int off = 128; off > 0; off >>= 1) {
      if (threadIdx.x < off) red[threadIdx.x] += red[threadIdx.x + off];
      __syncthreads();
    }
    col[e] = red[0];
    __syncthreads();
  }
  if (threadIdx.x == 0) {
    float mean = 0.25f * (col[0] + col[1] + col[2] + col[3]);
    float var = 0.f;
#pragma unroll
    for (int e = 0; e < NEXP; ++e) { float d = col[e] - mean; var += d * d; }
    var *= (1.f / 3.f);                       // ddof = 1
    out_loss[0] = var / (mean * mean);
  }
}

// ---------------------------------------------------------------------------
// Kernel 3 (fast path): BM=32 fused FFN, packed bf16 weights, dynamic LDS.
// One block = 32 tokens (two 16-row sub-tiles), 8 waves.  Each B fragment
// loaded once feeds TWO WMMAs -> halves L2 weight traffic.
// h1 is staged in LDS already in WMMA A-fragment order, so phase-2 A loads
// are 2x ds_load_b128 from a single base address.
// ---------------------------------------------------------------------------
#define SM_XA   0                                    // [2][32][32][8] u32  64KB
#define SM_H1   (SM_XA + 2 * 32 * 32 * 8 * 4)        // [8][2][32][8] u32   16KB
#define SM_CW   (SM_H1 + 8 * 2 * 32 * 8 * 4)         // [4][32] f32        512B
#define SM_EACT (SM_CW + NEXP * 32 * 4)              // [4] int
#define SM_TOTAL (SM_EACT + NEXP * 4)

__global__ __launch_bounds__(256, 1) void moe_ffn32(
    const float* __restrict__ x, const float* __restrict__ combine,
    const unsigned* __restrict__ W1p, const float* __restrict__ b1,
    const unsigned* __restrict__ W2p, const float* __restrict__ b2,
    float* __restrict__ out) {
  extern __shared__ __attribute__((aligned(128))) char smem[];
  unsigned (*xA)[32][32][8] = (unsigned(*)[32][32][8])
      __builtin_assume_aligned(smem + SM_XA, 128);
  // h1 fragment store: dword index (((j*2+sub)*32 + lane')*8 + v), halves=bf16
  unsigned* h1f = (unsigned*)__builtin_assume_aligned(smem + SM_H1, 128);
  unsigned short* h1h = (unsigned short*)h1f;
  float* cw = (float*)(smem + SM_CW);                // cw[e*32 + tok]
  int*   eact = (int*)(smem + SM_EACT);

  const int tid  = threadIdx.x;
  const int wave = tid >> 5;
  const int lane = tid & 31;
  const int tb   = blockIdx.x * 32;                  // token base
  const int m16  = lane & 15;
  const int hiHalf = (lane >= 16);

  if (tid < NEXP * 32) {
    int e = tid >> 5, m = tid & 31;
    cw[e * 32 + m] = combine[(size_t)(tb + m) * NEXP + e];
  }
  __syncthreads();
  if (tid < NEXP) {
    float s = 0.f;
    for (int m = 0; m < 32; ++m) s += fabsf(cw[tid * 32 + m]);
    eact[tid] = (s > 0.f);
  }
  // Pack the 32x1024 x tile into A-fragment layout (two 16-row sub-tiles).
  for (int i = tid; i < 2 * 32 * 32; i += 256) {
    int sub = i >> 10, kk = (i >> 5) & 31, ln = i & 31;
    const float* xr = x + (size_t)(tb + sub * 16 + (ln & 15)) * DIM + kk * 32;
    int kb = (ln < 16) ? 0 : 8;
#pragma unroll
    for (int v = 0; v < 8; ++v) {
      int k = kb + ((v & 4) << 2) + 2 * (v & 3);
      xA[sub][kk][ln][v] = pack_bf16x2(xr[k], xr[k + 1]);
    }
  }
  __syncthreads();

  v8f acc[2][8] = {};                                // 32 tok x 128 d-cols

  // h1 fragment-store coordinates (constant per lane/t):
  // element (m, k=16t+m16): reader lane' = m + (m16&8 ? 16 : 0),
  // vgpr v = 4t + ((m16&7)>>1), half = m16&1.
  const int lpOff = (m16 & 8) ? 16 : 0;
  const int vLow  = (m16 & 7) >> 1;
  const int half  = m16 & 1;

  for (int e = 0; e < NEXP; ++e) {
    if (!eact[e]) continue;                          // block-uniform skip

    for (int hc = 0; hc < HID; hc += 256) {
      __syncthreads();                               // h1f reuse guard
      // ---- phase 1: two h-tiles per wave, B shared across both sub-tiles
#pragma unroll
      for (int t = 0; t < 2; ++t) {
        const int hb = hc + 32 * wave + 16 * t;
        v8f c0 = {}, c1 = {};
        const unsigned* bp = W1p +
            ((size_t)(e * (HID / 16) + (hb >> 4)) * 32 * 32 + lane) * 8;
        for (int kk = 0; kk < 32; ++kk, bp += 32 * 8) {
          BF16Frag a0, a1, b;
          __builtin_prefetch(bp + 2 * 32 * 8, 0, 3);
#pragma unroll
          for (int v = 0; v < 8; ++v) b.u[v] = bp[v];
#pragma unroll
          for (int v = 0; v < 8; ++v) a0.u[v] = xA[0][kk][lane][v];
#pragma unroll
          for (int v = 0; v < 8; ++v) a1.u[v] = xA[1][kk][lane][v];
          c0 = __builtin_amdgcn_wmma_f32_16x16x32_bf16(
                   false, a0.v, false, b.v, (short)0, c0, false, false);
          c1 = __builtin_amdgcn_wmma_f32_16x16x32_bf16(
                   false, a1.v, false, b.v, (short)0, c1, false, false);
        }
        const float bias = b1[e * HID + hb + m16];
        const int vIdx = 4 * t + vLow;
#pragma unroll
        for (int r = 0; r < 8; ++r) {
          int mloc = r + (hiHalf ? 8 : 0);
          float h0 = c0[r] + bias, h1 = c1[r] + bias;
          h0 = h0 > 0.f ? h0 : 0.f;
          h1 = h1 > 0.f ? h1 : 0.f;
          h0 *= cw[e * 32 + mloc];                   // pre-scale (GEMM2 linear)
          h1 *= cw[e * 32 + 16 + mloc];
          int lp = mloc + lpOff;
          int u0 = (((wave * 2 + 0) * 32 + lp) * 8 + vIdx) * 2 + half;
          h1h[u0]             = f32_bf16(h0);
          h1h[u0 + 32 * 8 * 2] = f32_bf16(h1);       // sub-tile 1 stride
        }
      }
      __syncthreads();
      // ---- phase 2: acc += h1_chunk @ W2e^T, B shared across sub-tiles
      for (int j = 0; j < 8; ++j) {
        BF16Frag a0, a1;
        const unsigned* ap = h1f + ((j * 2) * 32 + lane) * 8;
#pragma unroll
        for (int v = 0; v < 8; ++v) a0.u[v] = ap[v];
#pragma unroll
        for (int v = 0; v < 8; ++v) a1.u[v] = ap[32 * 8 + v];
        const int kk2 = (hc >> 5) + j;
#pragma unroll
        for (int dt = 0; dt < 8; ++dt) {
          const int rt = e * (DIM / 16) + 8 * wave + dt;
          const unsigned* bp = W2p +
              (((size_t)rt * (HID / 32) + kk2) * 32 + lane) * 8;
          BF16Frag b;
#pragma unroll
          for (int v = 0; v < 8; ++v) b.u[v] = bp[v];
          acc[0][dt] = __builtin_amdgcn_wmma_f32_16x16x32_bf16(
                           false, a0.v, false, b.v, (short)0, acc[0][dt], false, false);
          acc[1][dt] = __builtin_amdgcn_wmma_f32_16x16x32_bf16(
                           false, a1.v, false, b.v, (short)0, acc[1][dt], false, false);
        }
      }
    }
  }

  // Epilogue: C/D layout -> global, plus sum_e cw*b2 bias term.
#pragma unroll
  for (int s = 0; s < 2; ++s) {
#pragma unroll
    for (int dt = 0; dt < 8; ++dt) {
      const int dcol = 128 * wave + 16 * dt + m16;
#pragma unroll
      for (int r = 0; r < 8; ++r) {
        int mloc = s * 16 + r + (hiHalf ? 8 : 0);
        float bias = 0.f;
#pragma unroll
        for (int e = 0; e < NEXP; ++e) bias += cw[e * 32 + mloc] * b2[e * DIM + dcol];
        out[(size_t)(tb + mloc) * DIM + dcol] = acc[s][dt][r] + bias;
      }
    }
  }
}

// ---------------------------------------------------------------------------
// Fallback (small workspace): BM=16, f32 weights converted inline.
// ---------------------------------------------------------------------------
__global__ __launch_bounds__(256) void moe_ffn16_direct(
    const float* __restrict__ x, const float* __restrict__ combine,
    const float* __restrict__ W1, const float* __restrict__ b1,
    const float* __restrict__ W2, const float* __restrict__ b2,
    float* __restrict__ out) {
  __shared__ unsigned xA[32][32][8];
  __shared__ unsigned short h1s[8][16][32];
  __shared__ float cw[NEXP][16];
  __shared__ int eact[NEXP];

  const int tid  = threadIdx.x;
  const int wave = tid >> 5;
  const int lane = tid & 31;
  const int tb   = blockIdx.x * 16;
  const int m16  = lane & 15;
  const int hiHalf = (lane >= 16);

  if (tid < NEXP * 16) {
    int e = tid >> 4, m = tid & 15;
    cw[e][m] = combine[(size_t)(tb + m) * NEXP + e];
  }
  __syncthreads();
  if (tid < NEXP) {
    float s = 0.f;
    for (int m = 0; m < 16; ++m) s += fabsf(cw[tid][m]);
    eact[tid] = (s > 0.f);
  }
  for (int i = tid; i < 32 * 32; i += 256) {
    int kk = i >> 5, ln = i & 31;
    const float* xr = x + (size_t)(tb + (ln & 15)) * DIM + kk * 32;
    int kb = (ln < 16) ? 0 : 8;
#pragma unroll
    for (int v = 0; v < 8; ++v) {
      int k = kb + ((v & 4) << 2) + 2 * (v & 3);
      xA[kk][ln][v] = pack_bf16x2(xr[k], xr[k + 1]);
    }
  }
  __syncthreads();

  v8f acc[8] = {};

  for (int e = 0; e < NEXP; ++e) {
    if (!eact[e]) continue;
    const float* W1e = W1 + (size_t)e * HID * DIM;
    const float* W2e = W2 + (size_t)e * DIM * HID;

    for (int hc = 0; hc < HID; hc += 256) {
      __syncthreads();
#pragma unroll
      for (int t = 0; t < 2; ++t) {
        const int hb = hc + 32 * wave + 16 * t;
        v8f c = {};
        const float* wrow = W1e + (size_t)(hb + m16) * DIM + (hiHalf ? 8 : 0);
        for (int kk = 0; kk < 32; ++kk) {
          BF16Frag a, b;
#pragma unroll
          for (int v = 0; v < 8; ++v) a.u[v] = xA[kk][lane][v];
          const float* wr = wrow + kk * 32;
#pragma unroll
          for (int v = 0; v < 8; ++v) {
            float2 f = *(const float2*)(wr + ((v & 4) << 2) + 2 * (v & 3));
            b.u[v] = pack_bf16x2(f.x, f.y);
          }
          c = __builtin_amdgcn_wmma_f32_16x16x32_bf16(
                  false, a.v, false, b.v, (short)0, c, false, false);
        }
        const float bias = b1[e * HID + hb + m16];
#pragma unroll
        for (int r = 0; r < 8; ++r) {
          int m = r + (hiHalf ? 8 : 0);
          float hv = c[r] + bias;
          hv = hv > 0.f ? hv : 0.f;
          hv *= cw[e][m];
          h1s[wave][m][16 * t + m16] = f32_bf16(hv);
        }
      }
      __syncthreads();
      for (int j = 0; j < 8; ++j) {
        BF16Frag a;
        const int kb = hiHalf ? 8 : 0;
#pragma unroll
        for (int v = 0; v < 8; ++v) {
          int k = kb + ((v & 4) << 2) + 2 * (v & 3);
          a.u[v] = *(const unsigned*)&h1s[j][m16][k];
        }
#pragma unroll
        for (int dt = 0; dt < 8; ++dt) {
          const int dcol = 128 * wave + 16 * dt + m16;
          const int hb = hc + 32 * j + (hiHalf ? 8 : 0);
          const float* wr = W2e + (size_t)dcol * HID + hb;
          BF16Frag b;
#pragma unroll
          for (int v = 0; v < 8; ++v) {
            float2 f = *(const float2*)(wr + ((v & 4) << 2) + 2 * (v & 3));
            b.u[v] = pack_bf16x2(f.x, f.y);
          }
          acc[dt] = __builtin_amdgcn_wmma_f32_16x16x32_bf16(
                        false, a.v, false, b.v, (short)0, acc[dt], false, false);
        }
      }
    }
  }

#pragma unroll
  for (int dt = 0; dt < 8; ++dt) {
    const int dcol = 128 * wave + 16 * dt + m16;
#pragma unroll
    for (int r = 0; r < 8; ++r) {
      int m = r + (hiHalf ? 8 : 0);
      float bias = 0.f;
#pragma unroll
      for (int e = 0; e < NEXP; ++e) bias += cw[e][m] * b2[e * DIM + dcol];
      out[(size_t)(tb + m) * DIM + dcol] = acc[dt][r] + bias;
    }
  }
}

// ---------------------------------------------------------------------------
extern "C" void kernel_launch(void* const* d_in, const int* in_sizes, int n_in,
                              void* d_out, int out_size, void* d_ws, size_t ws_size,
                              hipStream_t stream) {
  (void)in_sizes; (void)n_in; (void)out_size;
  const float* x   = (const float*)d_in[0];
  const float* eps = (const float*)d_in[1];
  const float* Wg  = (const float*)d_in[2];
  const float* Wn  = (const float*)d_in[3];
  const float* W1  = (const float*)d_in[4];
  const float* b1  = (const float*)d_in[5];
  const float* W2  = (const float*)d_in[6];
  const float* b2  = (const float*)d_in[7];
  // d_in[8] = k (fixed at 2, baked into the gating kernel)

  float* out     = (float*)d_out;                 // [N*D] out, then [1] loss
  float* combine = (float*)d_ws;                  // [N, 4] routing weights
  const size_t combBytes = (size_t)N_TOK * NEXP * sizeof(float);     // 128 KB

  const size_t w1pDw = (size_t)(NEXP * HID / 16) * (DIM / 32) * 32 * 8;
  const size_t w2pDw = (size_t)(NEXP * DIM / 16) * (HID / 32) * 32 * 8;
  const size_t need  = combBytes + (w1pDw + w2pDw) * sizeof(unsigned);

  moe_gate<<<N_TOK / 8, 256, 0, stream>>>(x, eps, Wg, Wn, combine);
  moe_loss<<<1, 256, 0, stream>>>(combine, out + (size_t)N_TOK * DIM);

  if (ws_size >= need) {
    unsigned* W1p = (unsigned*)((char*)d_ws + combBytes);
    unsigned* W2p = W1p + w1pDw;
    moe_pack_frag<<<(int)((w1pDw + 255) / 256), 256, 0, stream>>>(
        W1, W1p, DIM / 32, (long)w1pDw);
    moe_pack_frag<<<(int)((w2pDw + 255) / 256), 256, 0, stream>>>(
        W2, W2p, HID / 32, (long)w2pDw);
    moe_ffn32<<<N_TOK / 32, 256, SM_TOTAL, stream>>>(
        x, combine, W1p, b1, W2p, b2, out);
  } else {
    moe_ffn16_direct<<<N_TOK / 16, 256, 0, stream>>>(
        x, combine, W1, b1, W2, b2, out);
  }
}
